// ScaledDotProductAttention_16157666967925
// MI455X (gfx1250) — matrix operand order repaired and checked by reading, not compiled, not run on hardware
//
#include <hip/hip_runtime.h>

// ---- CDNA5 WMMA types ----
typedef __attribute__((ext_vector_type(16))) __bf16 v16bf;
typedef __attribute__((ext_vector_type(8)))  float  v8f;

// Problem constants (from reference: B=4,H=16,S=2048,D=64)
constexpr int B_ = 4, H_ = 16, S_ = 2048, D_ = 64;
constexpr int QT = 32;            // query rows per workgroup (2 WMMA M-tiles)
constexpr int NW = 8;             // waves per workgroup (wave32)
constexpr int NTHREADS = NW * 32; // 256
constexpr float SCALE_INV = 0.125f;       // 1/sqrt(64)
constexpr float NEG_BIG = -100000000.0f;  // matches reference NEG_INF

// fp32 -> bf16 using the native gfx1250 convert path (v_cvt_*_bf16_f32),
// instead of manual integer RNE (which cost ~4 VALU ops/element in round 1).
static __device__ __forceinline__ __bf16 f2bf(float f) {
    return (__bf16)f;
}

__global__ __launch_bounds__(NTHREADS, 1)
void attn_fused_kernel(const float* __restrict__ Q,
                       const float* __restrict__ K,
                       const float* __restrict__ V,
                       const unsigned char* __restrict__ mask,
                       float* __restrict__ out_attn,
                       float* __restrict__ out_w) {
    // Dynamic LDS partition:
    //   lds_s : QT * S   fp32 scores/probs   (256 KB)  -- needs CDNA5's 320KB WGP LDS
    //   lds_o : QT * D   fp32 PV accumulator (8 KB)
    //   red   : 256      fp32 row reductions
    //   rinv  : QT       fp32 1/rowsum
    extern __shared__ float smem[];
    float* lds_s = smem;
    float* lds_o = lds_s + QT * S_;
    float* red   = lds_o + QT * D_;
    float* rinv  = red + NTHREADS;

    const int tid  = threadIdx.x;
    const int lane = tid & 31;
    const int wave = tid >> 5;
    const int half = lane >> 4;   // 0 or 1: lane group
    const int ln   = lane & 15;

    const int bh = blockIdx.x / (S_ / QT);       // fused (b,h)
    const int q0 = (blockIdx.x % (S_ / QT)) * QT;

    // zero the cross-wave PV accumulator
    for (int i = tid; i < QT * D_; i += NTHREADS) lds_o[i] = 0.0f;

    // ---------------- load Q tile as bf16 A-fragments ----------------
    // A 16x32 bf16 layout: lanes0-15 & 16-31 both hold M=ln;
    // elems 0..7 -> K = 8*half+e ; elems 8..15 -> K = 16 + 8*half + (e-8)
    v16bf qf[2][2];
    #pragma unroll
    for (int mt = 0; mt < 2; ++mt) {
        const float* qrow = Q + ((size_t)bh * S_ + q0 + mt * 16 + ln) * D_;
        #pragma unroll
        for (int dk = 0; dk < 2; ++dk) {
            const int d0 = dk * 32;
            #pragma unroll
            for (int e = 0; e < 8; ++e)
                qf[mt][dk][e] = f2bf(qrow[d0 + 8 * half + e]);
            #pragma unroll
            for (int e = 8; e < 16; ++e)
                qf[mt][dk][e] = f2bf(qrow[d0 + 16 + 8 * half + (e - 8)]);
        }
    }

    // ---------------- S = Q K^T / sqrt(d), masked, into LDS ----------------
    // B 32x16 bf16 layout: lane holds N=ln, elems 0..7 -> K = 8*half+e ; elems 8..15 -> K = 16 + 8*half + (e-8)
    for (int kc = wave * 16; kc < S_; kc += NW * 16) {
        // prefetch next chunk of K rows for this wave (global_prefetch_b8)
        if (kc + NW * 16 < S_) {
            __builtin_prefetch(K + ((size_t)bh * S_ + kc + NW * 16 + ln) * D_, 0, 1);
        }
        v16bf kf[2];
        #pragma unroll
        for (int dk = 0; dk < 2; ++dk) {
            const float* kb = K + ((size_t)bh * S_ + kc + ln) * D_ + dk * 32 + 8 * half;
            #pragma unroll
            for (int e = 0; e < 16; ++e) kf[dk][e] = f2bf(kb[e < 8 ? e : e + 8]);
        }
        #pragma unroll
        for (int mt = 0; mt < 2; ++mt) {
            v8f acc = {};
            acc = __builtin_amdgcn_wmma_f32_16x16x32_bf16(false, qf[mt][0], false, kf[0],
                                                          (short)0, acc, false, false);
            acc = __builtin_amdgcn_wmma_f32_16x16x32_bf16(false, qf[mt][1], false, kf[1],
                                                          (short)0, acc, false, false);
            // C/D layout: vgpr r, M = 8*half + r, N = ln
            #pragma unroll
            for (int r = 0; r < 8; ++r) {
                const int m   = mt * 16 + half * 8 + r;
                const int col = kc + ln;
                const float s = acc[r] * SCALE_INV;
                const unsigned char mk = mask[(size_t)(q0 + m) * S_ + col];
                lds_s[m * S_ + col] = mk ? s : NEG_BIG;
            }
        }
    }
    __syncthreads();

    // ---------------- row softmax over S=2048 (8 threads / row) ----------------
    const int row = tid >> 3, seg = tid & 7;
    float* srow = lds_s + row * S_ + seg * (S_ / 8);
    float lmax = -3.4e38f;
    for (int j = 0; j < S_ / 8; ++j) lmax = fmaxf(lmax, srow[j]);
    red[tid] = lmax;
    __syncthreads();
    float rmax = red[row << 3];
    #pragma unroll
    for (int t = 1; t < 8; ++t) rmax = fmaxf(rmax, red[(row << 3) + t]);
    __syncthreads();
    float lsum = 0.0f;
    for (int j = 0; j < S_ / 8; ++j) {
        float e = __expf(srow[j] - rmax);
        srow[j] = e;
        lsum += e;
    }
    red[tid] = lsum;
    __syncthreads();
    float tot = red[row << 3];
    #pragma unroll
    for (int t = 1; t < 8; ++t) tot += red[(row << 3) + t];
    if (seg == 0) rinv[row] = 1.0f / tot;
    __syncthreads();

    // ---------------- stream normalized weights to global (float4) ----------------
    {
        float4* wdst = (float4*)(out_w + ((size_t)bh * S_ + q0) * S_);
        const float4* wsrc = (const float4*)lds_s;
        for (int i = tid; i < QT * S_ / 4; i += NTHREADS) {
            float4 v = wsrc[i];
            const float s = rinv[i >> 9];   // 512 float4 per row
            v.x *= s; v.y *= s; v.z *= s; v.w *= s;
            wdst[i] = v;
        }
    }

    // ---------------- O = P V  (unnormalized P from LDS; fold 1/sum at the end) ----
    v8f pacc[2][4];
    #pragma unroll
    for (int mt = 0; mt < 2; ++mt)
        #pragma unroll
        for (int nt = 0; nt < 4; ++nt) { v8f z = {}; pacc[mt][nt] = z; }

    for (int kc = wave * 32; kc < S_; kc += NW * 32) {
        // prefetch this wave's next V chunk
        if (kc + NW * 32 < S_) {
            __builtin_prefetch(V + ((size_t)bh * S_ + kc + NW * 32 + 16 * half) * D_, 0, 1);
        }
        v16bf pf[2];
        #pragma unroll
        for (int mt = 0; mt < 2; ++mt) {
            const float* pr = lds_s + (mt * 16 + ln) * S_ + kc;
            #pragma unroll
            for (int e = 0; e < 8; ++e)
                pf[mt][e] = f2bf(pr[8 * half + e]);
            #pragma unroll
            for (int e = 8; e < 16; ++e)
                pf[mt][e] = f2bf(pr[16 + 8 * half + (e - 8)]);
        }
        #pragma unroll
        for (int nt = 0; nt < 4; ++nt) {
            v16bf vf;
            const float* vb = V + ((size_t)bh * S_ + kc + 8 * half) * D_ + nt * 16 + ln;
            #pragma unroll
            for (int e = 0; e < 16; ++e) vf[e] = f2bf(vb[(size_t)(e < 8 ? e : e + 8) * D_]);
            #pragma unroll
            for (int mt = 0; mt < 2; ++mt)
                pacc[mt][nt] = __builtin_amdgcn_wmma_f32_16x16x32_bf16(
                    false, pf[mt], false, vf, (short)0, pacc[mt][nt], false, false);
        }
    }

    // cross-wave reduction into LDS via ds_add_f32 atomics
    #pragma unroll
    for (int mt = 0; mt < 2; ++mt)
        #pragma unroll
        for (int nt = 0; nt < 4; ++nt)
            #pragma unroll
            for (int r = 0; r < 8; ++r)
                atomicAdd(&lds_o[(mt * 16 + half * 8 + r) * D_ + nt * 16 + ln],
                          pacc[mt][nt][r]);
    __syncthreads();

    // final normalized output
    for (int i = tid; i < QT * D_; i += NTHREADS) {
        const int m = i >> 6;
        out_attn[((size_t)bh * S_ + q0 + m) * D_ + (i & 63)] = lds_o[i] * rinv[m];
    }
}

extern "C" void kernel_launch(void* const* d_in, const int* in_sizes, int n_in,
                              void* d_out, int out_size, void* d_ws, size_t ws_size,
                              hipStream_t stream) {
    (void)in_sizes; (void)n_in; (void)d_ws; (void)ws_size; (void)out_size;

    const float* Q = (const float*)d_in[0];
    const float* K = (const float*)d_in[1];
    const float* V = (const float*)d_in[2];
    const unsigned char* mask = (const unsigned char*)d_in[3];

    float* out_attn = (float*)d_out;                                   // [B,H,S,D]
    float* out_w    = out_attn + (size_t)B_ * H_ * S_ * D_;            // [B,H,S,S]

    const size_t smem_bytes =
        (size_t)(QT * S_ + QT * D_ + NTHREADS + QT) * sizeof(float);   // ~265 KB (<320KB WGP LDS)

    const int grid = B_ * H_ * (S_ / QT);   // 4096 workgroups
    attn_fused_kernel<<<grid, NTHREADS, smem_bytes, stream>>>(Q, K, V, mask,
                                                              out_attn, out_w);
}
